// TorchBernoulliLayer_44057774522845
// MI455X (gfx1250) — matrix operand order repaired and checked by reading, not compiled, not run on hardware
//
#include <hip/hip_runtime.h>
#include <cstdint>

typedef __attribute__((ext_vector_type(16))) _Float16 v16h;
typedef __attribute__((ext_vector_type(8)))  float    v8f;

#define NN     8192
#define NBATCH 2048
#define ALPHA  0.05f
#define LAM    0.95f
#define EXPM05 0.60653065971f   // e^-0.5
#define CHUNK  512
#define HALO   384   // lam^384 ~ 3e-9: below f32 noise for the membrane state

// select b[s] from 8 register-resident values with a cndmask tree (s = 0..7)
__device__ __forceinline__ uint32_t sel8(const uint32_t b[8], int s) {
  uint32_t t0 = (s & 1) ? b[1] : b[0];
  uint32_t t1 = (s & 1) ? b[3] : b[2];
  uint32_t t2 = (s & 1) ? b[5] : b[4];
  uint32_t t3 = (s & 1) ? b[7] : b[6];
  uint32_t u0 = (s & 2) ? t1 : t0;
  uint32_t u1 = (s & 2) ? t3 : t2;
  return (s & 4) ? u1 : u0;
}

// spike candidates: u < sigmoid(v)  <=>  u*(1+e^-v) < 1  (one exp, no logs)
__device__ __forceinline__ void spike2(float v, float u, bool& s0, bool& s1) {
  float e = __expf(-v);
  s0 = __builtin_fmaf(u, e, u) < 1.0f;            // u < sigmoid(v)
  s1 = __builtin_fmaf(u, e * EXPM05, u) < 1.0f;   // u < sigmoid(v + 0.5)
}

// Kernel 1: WMMA membrane-potential scan. Per 32-row superblock:
//   rows 0..15:  D1 = A1(16x32,lower-tri) @ B + lam^(r+1) * carry
//   rows 16..31: D2 = A2(16x32,full)      @ B + lam^(r+17) * carry
// Candidate spike bits packed via wave32 ballots into per-(row, 16-col tile)
// words: s0_16 | s1_16<<16.
__global__ void lif_membrane_bits(const float* __restrict__ I,
                                  const float* __restrict__ U,
                                  const float* __restrict__ V0,
                                  uint32_t* __restrict__ S) {
  const int lane = threadIdx.x & 31;
  // force wave-uniform values into SGPRs: scalar loop control, EXEC untouched
  const int wvid = __builtin_amdgcn_readfirstlane(
      (int)((blockIdx.x * blockDim.x + threadIdx.x) >> 5));
  const int t     = wvid & 511;     // 16-neuron column tile (512 tiles)
  const int chunk = wvid >> 9;      // batch chunk 0..3 (512 rows each)
  const int cl    = lane & 15;
  const int half  = lane >> 4;
  const int col   = t * 16 + cl;

  // A matrices (f16), A-layout: K(h) = h + 8*((h>=8)+half), row M = cl.
  v16h a1, a2;
  const int m = cl;
  #pragma unroll
  for (int h = 0; h < 16; ++h) {
    int j = h + 8 * ((h >> 3) + half);
    float v1 = (j <= m)      ? ALPHA * __powf(LAM, (float)(m - j))      : 0.0f;
    float v2 = (j <= m + 16) ? ALPHA * __powf(LAM, (float)(m + 16 - j)) : 0.0f;
    a1[h] = (_Float16)v1;
    a2[h] = (_Float16)v2;
  }
  // carry decay factors for C accumulator (D/C layout: vgpr i <-> row i+8*half)
  float f1[8], f2[8];
  #pragma unroll
  for (int i = 0; i < 8; ++i) {
    int r = i + 8 * half;
    f1[i] = __powf(LAM, (float)(r + 1));
    f2[i] = f1[i] * __powf(LAM, 16.0f);
  }

  const int cstart = chunk * CHUNK;                   // scalar
  const int hstart = chunk ? (cstart - HALO) : 0;     // scalar
  const int nsb    = (cstart + CHUNK - hstart) >> 5;  // scalar: 16 or 28

  // carry = membrane potential just before hstart (exact for chunk 0)
  float c = V0[col] * __powf(LAM, (float)hstart);

  // Base pointers, advanced once per superblock; all loads/stores below use
  // compile-time immediate offsets (max ~1.8MB < 24-bit signed IOFFSET).
  const float* pI = I + (size_t)(hstart + 8 * half) * NN + col;
  const float* pU = U + (size_t)(hstart + 8 * half) * NN + col;
  uint32_t*    pS = S + ((size_t)hstart + lane) * 512 + t;

  // B tile (32x16 f16): lane holds K = h + 8*((h>=8)+half) -> rows b0+K
  v16h bcur;
  #pragma unroll
  for (int h = 0; h < 16; ++h)
    bcur[h] = (_Float16)pI[(h + 8 * (h >> 3)) * NN];

  for (int sb = 0; sb < nsb; ++sb) {
    const bool emit = (sb >= nsb - 16);  // scalar: last 16 superblocks = body

    // prefetch next superblock's B tile (immediate offsets off same base)
    v16h bnext = bcur;
    if (sb + 1 < nsb) {
      #pragma unroll
      for (int h = 0; h < 16; ++h)
        bnext[h] = (_Float16)pI[(32 + h + 8 * (h >> 3)) * NN];
    }

    // rows 16..31 of superblock (also produces next carry)
    v8f c2;
    #pragma unroll
    for (int i = 0; i < 8; ++i) c2[i] = f2[i] * c;
    v8f d2 = __builtin_amdgcn_wmma_f32_16x16x32_f16(false, a2, false, bcur,
                                                    (short)0, c2, false, false);

    if (emit) {
      v8f c1;
      #pragma unroll
      for (int i = 0; i < 8; ++i) c1[i] = f1[i] * c;
      v8f d1 = __builtin_amdgcn_wmma_f32_16x16x32_f16(false, a1, false, bcur,
                                                      (short)0, c1, false, false);

      // candidate spike bits + ballots; one ballot covers 2 rows (lo/hi lanes)
      uint32_t bal0[2][8], bal1[2][8];
      #pragma unroll
      for (int i = 0; i < 8; ++i) {
        bool s0, s1;
        spike2(d1[i], pU[i * NN], s0, s1);
        bal0[0][i] = __builtin_amdgcn_ballot_w32(s0);
        bal1[0][i] = __builtin_amdgcn_ballot_w32(s1);
        spike2(d2[i], pU[(16 + i) * NN], s0, s1);
        bal0[1][i] = __builtin_amdgcn_ballot_w32(s0);
        bal1[1][i] = __builtin_amdgcn_ballot_w32(s1);
      }
      // lane j (0..15) stores the word for row b0 + 16*w + j
      #pragma unroll
      for (int w = 0; w < 2; ++w) {
        uint32_t x0 = sel8(bal0[w], lane & 7);
        uint32_t x1 = sel8(bal1[w], lane & 7);
        uint32_t word = (lane & 8) ? ((x0 >> 16) | (x1 & 0xFFFF0000u))
                                   : ((x0 & 0xFFFFu) | (x1 << 16));
        if (lane < 16)
          pS[w * 16 * 512] = word;   // row = b0 + 16*w + lane, tile t
      }
    }

    // next carry = v at row b0+31 = D2 row 15 (vgpr7, upper lanes)
    float v7 = d2[7];
    float ov = __shfl_xor(v7, 16, 32);
    c = half ? v7 : ov;

    bcur = bnext;
    pI += (size_t)32 * NN;
    pU += (size_t)32 * NN;
    pS += (size_t)32 * 512;
  }
}

// Kernel 2: resolve the autoregressive spike chain per row. Transition maps are
// monotone (const0/identity/const1). Bit-parallel via addition carry chain:
// g = D (const-1 positions), p = ~C (identity positions); out = carry vector of
// (D | ~C) + D + cin. One wave per row; ballot scan for cross-lane carry.
__global__ void lif_spike_resolve(const uint32_t* __restrict__ S,
                                  float* __restrict__ out) {
  const int lane = threadIdx.x & 31;  // owns col tiles [16*lane, 16*lane+16)
  const int r = __builtin_amdgcn_readfirstlane(
      (int)((blockIdx.x * blockDim.x + threadIdx.x) >> 5));  // row (scalar)

  const uint32_t* pS = S + (size_t)r * 512 + lane * 16;
  uint32_t w[16];
  #pragma unroll
  for (int i = 0; i < 16; ++i) w[i] = pS[i];

  // phase 1: local chain with cin=0; summary = (anyConst, carry-out)
  uint32_t cin = 0;
  bool any = false;
  #pragma unroll
  for (int i = 0; i < 16; ++i) {
    uint32_t s0 = w[i] & 0xFFFFu, s1 = w[i] >> 16;
    uint32_t C = (~(s0 ^ s1)) & 0xFFFFu;   // constant positions
    uint32_t D = s0 & s1;                  // const-1 positions
    uint32_t A = D | ((~C) & 0xFFFFu);
    uint32_t sum = A + D + cin;
    uint32_t cv  = sum ^ A ^ D;
    uint32_t o   = (cv >> 1) & 0xFFFFu;
    cin = o >> 15;
    any = any || (C != 0u);
  }
  uint32_t Cm = __builtin_amdgcn_ballot_w32(any);
  uint32_t Vm = __builtin_amdgcn_ballot_w32(any && (cin != 0u));
  uint32_t below = Cm & ((1u << lane) - 1u);
  uint32_t cin2 = below ? ((Vm >> (31 - __clz((int)below))) & 1u) : 0u;

  // phase 2: re-run with true carry-in, expand bits to f32 spikes
  float* po = out + (size_t)r * NN + lane * 256;
  #pragma unroll
  for (int i = 0; i < 16; ++i) {
    uint32_t s0 = w[i] & 0xFFFFu, s1 = w[i] >> 16;
    uint32_t C = (~(s0 ^ s1)) & 0xFFFFu;
    uint32_t D = s0 & s1;
    uint32_t A = D | ((~C) & 0xFFFFu);
    uint32_t sum = A + D + cin2;
    uint32_t cv  = sum ^ A ^ D;
    uint32_t o   = (cv >> 1) & 0xFFFFu;
    cin2 = o >> 15;
    #pragma unroll
    for (int g = 0; g < 4; ++g) {
      float4 f;
      f.x = (float)((o >> (4 * g + 0)) & 1u);
      f.y = (float)((o >> (4 * g + 1)) & 1u);
      f.z = (float)((o >> (4 * g + 2)) & 1u);
      f.w = (float)((o >> (4 * g + 3)) & 1u);
      *reinterpret_cast<float4*>(po + i * 16 + 4 * g) = f;
    }
  }
}

extern "C" void kernel_launch(void* const* d_in, const int* in_sizes, int n_in,
                              void* d_out, int out_size, void* d_ws, size_t ws_size,
                              hipStream_t stream) {
  const float* I  = (const float*)d_in[0];   // (2048, 8192) f32
  const float* U  = (const float*)d_in[1];   // (2048, 8192) f32
  const float* V0 = (const float*)d_in[2];   // (8192,) f32
  float* out = (float*)d_out;
  uint32_t* S = (uint32_t*)d_ws;             // needs 2048*512*4 = 4 MB scratch

  // K1: 512 col tiles x 4 batch chunks = 2048 waves (256 blocks x 256 thr)
  lif_membrane_bits<<<256, 256, 0, stream>>>(I, U, V0, S);
  // K2: one wave per row = 2048 waves
  lif_spike_resolve<<<256, 256, 0, stream>>>(S, out);
}
